// WAM_50483045597849
// MI455X (gfx1250) — matrix operand order repaired and checked by reading, not compiled, not run on hardware
//
#include <hip/hip_runtime.h>

// DANet dual-attention head for MI455X (gfx1250, wave32).
// All GEMM-shaped work on v_wmma_f32_16x16x32_f16; PAM as flash attention.
// Round 2: branchless flash operand loads via zero-padded q/k rows (d padded
// 8->16 in memory, 16->32 with a constant-zero register half); GEMM M split
// across blockIdx.y for 2x wave count on conv5.

typedef _Float16 h8   __attribute__((ext_vector_type(8)));
typedef _Float16 v16h __attribute__((ext_vector_type(16)));
typedef float    v8f  __attribute__((ext_vector_type(8)));

#define WMMA_F16(a,b,c) __builtin_amdgcn_wmma_f32_16x16x32_f16(false,(a),false,(b),(short)0,(c),false,false)

__device__ __forceinline__ v16h ld2h(const _Float16* p0, const _Float16* p1){
  union { v16h v; h8 h[2]; } u;
  u.h[0] = *(const h8*)p0;
  u.h[1] = *(const h8*)p1;
  return u.v;
}
__device__ __forceinline__ v16h ldlowz(const _Float16* p, h8 z){
  union { v16h v; h8 h[2]; } u;
  u.h[0] = *(const h8*)p;
  u.h[1] = z;
  return u.v;
}
__device__ __forceinline__ v8f zero8(){
  v8f z;
#pragma unroll
  for (int i=0;i<8;++i) z[i]=0.f;
  return z;
}
__device__ __forceinline__ h8 zeroh8(){
  h8 z;
#pragma unroll
  for (int i=0;i<8;++i) z[i]=(_Float16)0.f;
  return z;
}

// ---------------- weight packing (f32 -> f16, row-major [M][K]) ----------------
__global__ void k_pack5(const float* __restrict__ wa, const float* __restrict__ wc,
                        _Float16* __restrict__ A){
  int i = blockIdx.x*blockDim.x + threadIdx.x;   // 128*2304
  int m = i / 2304, k = i - m*2304;
  float v = (m < 64) ? wa[m*2304 + k] : wc[(m-64)*2304 + k];
  A[i] = (_Float16)v;
}
__global__ void k_packf(const float* __restrict__ w, _Float16* __restrict__ A, int n){
  int i = blockIdx.x*blockDim.x + threadIdx.x;
  if (i < n) A[i] = (_Float16)w[i];
}

// ---------------- im2col (3x3, SAME) -> patches[n][k] f16, k=ci*9+kh*3+kw ------
__global__ void k_im2col(const float* __restrict__ X, _Float16* __restrict__ P,
                         int b, int Cin, int logC){
  int tid = blockIdx.x*blockDim.x + threadIdx.x;   // Cin*4096, ci minor
  int ci = tid & (Cin-1);
  int hw = tid >> logC;
  int h = hw >> 6, w = hw & 63;
  const float* xb = X + (((size_t)b*Cin + ci) << 12);
  _Float16* p = P + (size_t)hw*(size_t)(Cin*9) + (size_t)ci*9;
#pragma unroll
  for (int kh=0; kh<3; ++kh){
    int hy = h + kh - 1;
#pragma unroll
    for (int kw=0; kw<3; ++kw){
      int wx = w + kw - 1;
      float v = (hy>=0 && hy<64 && wx>=0 && wx<64) ? xb[(hy<<6)+wx] : 0.f;
      p[kh*3+kw] = (_Float16)v;
    }
  }
}

// ---------------- WMMA GEMM: C[m-slice x 4096] = A[M x K] * B^T ----------------
// B is patch-major [n][k].  Wave owns MT*16 rows starting at blockIdx.y*MT*16.
// MODE 0: stacked conv5a/5c -> rows<64: out1/bn1, rows>=64: out2/bn2 (BN+ReLU)
// MODE 1: out1 = relu(bn1(acc));  MODE 2: out1 += relu(bn1(acc))
template<int MT, int MODE>
__global__ void __launch_bounds__(128) k_gemm(
    const _Float16* __restrict__ A, const _Float16* __restrict__ Bp, int K, int b,
    float* __restrict__ out1, float* __restrict__ out2,
    const float* __restrict__ bn1, const float* __restrict__ bn2)
{
  const int lane = threadIdx.x & 31;
  const int wv   = threadIdx.x >> 5;
  const int l15  = lane & 15;
  const int hi   = lane >> 4;
  const int n0   = (blockIdx.x*4 + wv)*16;
  const int moff = blockIdx.y * (MT*16);

  v8f acc[MT];
#pragma unroll
  for (int t=0;t<MT;++t) acc[t] = zero8();

  const _Float16* brow = Bp + (size_t)(n0 + l15)*K + hi*16;
  const _Float16* abase = A + (size_t)moff*K;
  for (int k0=0; k0<K; k0+=32){
    v16h bop = ld2h(brow + k0, brow + k0 + 8);
#pragma unroll
    for (int t=0;t<MT;++t){
      const _Float16* ar = abase + (size_t)(t*16 + l15)*K + k0 + hi*16;
      v16h aop = ld2h(ar, ar + 8);
      acc[t] = WMMA_F16(aop, bop, acc[t]);
    }
  }

  const int nn = n0 + l15;
#pragma unroll
  for (int t=0;t<MT;++t){
    const int mbase = moff + t*16 + hi*8;
#pragma unroll
    for (int r=0;r<8;++r){
      int c = mbase + r;
      float v = acc[t][r];
      if (MODE == 0){
        const float* bn = (c < 64) ? bn1 : bn2;   // uniform per block (blockIdx.y)
        int cc = c & 63;
        float inv = bn[cc] * rsqrtf(bn[192+cc] + 1e-5f);
        float y = (v - bn[128+cc]) * inv + bn[64+cc];
        y = fmaxf(y, 0.f);
        float* o = (c < 64) ? out1 : out2;
        o[(((size_t)b*64 + cc) << 12) + nn] = y;
      } else {
        float inv = bn1[c] * rsqrtf(bn1[192+c] + 1e-5f);
        float y = (v - bn1[128+c]) * inv + bn1[64+c];
        y = fmaxf(y, 0.f);
        size_t idx = (((size_t)b*64 + c) << 12) + nn;
        if (MODE == 1) out1[idx] = y;
        else           out1[idx] += y;
      }
    }
  }
}

// ---------------- q/k/v 1x1 convs from feat1 (tiny) ---------------------------
// qT/kT rows padded to 16 halves: d 0..7 real, d 8..15 = 0 (branchless WMMA feed)
__global__ void k_qkv(const float* __restrict__ f1,
                      const float* __restrict__ qw, const float* __restrict__ qb,
                      const float* __restrict__ kw, const float* __restrict__ kb,
                      const float* __restrict__ vw, const float* __restrict__ vb,
                      _Float16* __restrict__ qT, _Float16* __restrict__ kT,
                      _Float16* __restrict__ V)
{
  int tid = blockIdx.x*blockDim.x + threadIdx.x;   // 16384 = B*N
  int b = tid >> 12, n = tid & 4095;
  const float* fb = f1 + ((size_t)b << 18) + n;
  float qa[8], ka[8], va[64];
#pragma unroll
  for (int d=0;d<8;++d){ qa[d]=0.f; ka[d]=0.f; }
#pragma unroll
  for (int d=0;d<64;++d) va[d]=0.f;
#pragma unroll 1
  for (int c=0;c<64;++c){
    float f = fb[(size_t)c << 12];
#pragma unroll
    for (int d=0;d<8;++d){ qa[d] += qw[d*64+c]*f; ka[d] += kw[d*64+c]*f; }
#pragma unroll
    for (int d=0;d<64;++d) va[d] += vw[d*64+c]*f;
  }
  _Float16* qrow = qT + ((size_t)tid << 4);
  _Float16* krow = kT + ((size_t)tid << 4);
#pragma unroll
  for (int d=0;d<8;++d){
    qrow[d]   = (_Float16)(qa[d]+qb[d]);
    qrow[d+8] = (_Float16)0.f;
    krow[d]   = (_Float16)(ka[d]+kb[d]);
    krow[d+8] = (_Float16)0.f;
  }
#pragma unroll
  for (int d=0;d<64;++d)
    V[(((size_t)b*64 + d) << 12) + n] = (_Float16)(va[d]+vb[d]);
}

// ---------------- PAM flash attention: one wave per 16-query block ------------
__global__ void __launch_bounds__(32) k_flash(
    const _Float16* __restrict__ qT, const _Float16* __restrict__ kT,
    const _Float16* __restrict__ V, const float* __restrict__ gp,
    float* __restrict__ SA)
{
  __shared__ __align__(16) _Float16 Pb[16*32];   // P tile, [m][n_local] halves
  __shared__ float sum16[16];
  const int lane = threadIdx.x;
  const int l15 = lane & 15;
  const int hi  = lane >> 4;
  const int b   = blockIdx.y;
  const int m0  = blockIdx.x * 16;
  const h8 zh = zeroh8();

  // Q operand: A[16 m x 32 dpad]; padded rows make this a plain b128 per lane.
  v16h qop = ldlowz(qT + ((size_t)(b*4096 + m0 + l15) << 4) + hi*8, zh);

  const _Float16* kb = kT + ((size_t)b << 16);
  float pmax[8];
#pragma unroll
  for (int r=0;r<8;++r) pmax[r] = -1e30f;

  // pass 1: row maxima (S recomputed with WMMA; S work is ~1 GF total)
  for (int nt=0; nt<256; ++nt){
    v16h kop = ldlowz(kb + ((size_t)(nt*16 + l15) << 4) + hi*8, zh);
    v8f s = WMMA_F16(qop, kop, zero8());
#pragma unroll
    for (int r=0;r<8;++r) pmax[r] = fmaxf(pmax[r], s[r]);
  }
#pragma unroll
  for (int off=8; off>=1; off>>=1)
#pragma unroll
    for (int r=0;r<8;++r)
      pmax[r] = fmaxf(pmax[r], __shfl_xor(pmax[r], off, 16));

  // pass 2: P = exp(S-max), accumulate row sums and P.V
  float psum[8];
#pragma unroll
  for (int r=0;r<8;++r) psum[r] = 0.f;
  v8f acc[4];
#pragma unroll
  for (int t=0;t<4;++t) acc[t] = zero8();

  const _Float16* vbase = V + ((size_t)b << 18);
  for (int nt=0; nt<128; ++nt){
    int nb = nt*32;
#pragma unroll
    for (int t=0;t<2;++t){
      v16h kop = ldlowz(kb + ((size_t)(nb + t*16 + l15) << 4) + hi*8, zh);
      v8f s = WMMA_F16(qop, kop, zero8());
      int ncol = t*16 + l15;
#pragma unroll
      for (int r=0;r<8;++r){
        float p = __expf(s[r] - pmax[r]);
        psum[r] += p;
        Pb[(hi*8 + r)*32 + ncol] = (_Float16)p;   // transpose via LDS
      }
    }
    __syncthreads();
    // B operand = P^T [32 n x 16 m]: lane = column m, halves = its n range
    v16h pop = ld2h(&Pb[l15*32 + hi*16], &Pb[l15*32 + hi*16 + 8]);
#pragma unroll
    for (int t=0;t<4;++t){
      const _Float16* vr = vbase + (((size_t)(t*16 + l15)) << 12) + nb + hi*16;
      v16h vop = ld2h(vr, vr + 8);
      acc[t] = WMMA_F16(vop, pop, acc[t]);
    }
    __syncthreads();
  }

#pragma unroll
  for (int off=8; off>=1; off>>=1)
#pragma unroll
    for (int r=0;r<8;++r)
      psum[r] += __shfl_xor(psum[r], off, 16);
  if (l15 == 0){
#pragma unroll
    for (int r=0;r<8;++r) sum16[hi*8 + r] = psum[r];
  }
  __syncthreads();
  float scale = gp[0] / sum16[l15];
  int m = m0 + l15;
#pragma unroll
  for (int t=0;t<4;++t)
#pragma unroll
    for (int r=0;r<8;++r){
      int c = t*16 + hi*8 + r;
      SA[(((size_t)b*64 + c) << 12) + m] = acc[t][r]*scale;
    }
}

// ---------------- CAM: channel attention (tiny, VALU) -------------------------
__global__ void k_cam_mean(const float* __restrict__ f2, float* __restrict__ mean){
  __shared__ float red[256];
  int bc = blockIdx.x;
  const float* row = f2 + ((size_t)bc << 12);
  float s = 0.f;
  for (int i=threadIdx.x; i<4096; i+=256) s += row[i];
  red[threadIdx.x] = s;
  __syncthreads();
  for (int o=128; o>0; o>>=1){
    if ((int)threadIdx.x < o) red[threadIdx.x] += red[threadIdx.x + o];
    __syncthreads();
  }
  if (threadIdx.x == 0) mean[bc] = red[0] * (1.f/4096.f);
}

__global__ void __launch_bounds__(32) k_cam_energy(const float* __restrict__ f2,
    const float* __restrict__ mean, float* __restrict__ attn){
  int bc = blockIdx.x;                 // b*64 + c
  int b = bc >> 6;
  int lane = threadIdx.x;
  const float* base = f2 + ((size_t)b << 18);
  float mc = mean[bc];
  float m1 = mean[(b<<6) + lane];
  float m2 = mean[(b<<6) + lane + 32];
  const float* rc = f2 + ((size_t)bc << 12);
  const float* r1 = base + ((size_t)lane << 12);
  const float* r2 = base + ((size_t)(lane+32) << 12);
  float a1=0.f, a2=0.f;
  for (int n=0;n<4096;++n){
    float fc = rc[n] - mc;
    a1 += fc * (r1[n] - m1);
    a2 += fc * (r2[n] - m2);
  }
  a1 *= (1.f/4096.f); a2 *= (1.f/4096.f);
  float mx = fmaxf(a1, a2);
  for (int off=16; off>=1; off>>=1) mx = fmaxf(mx, __shfl_xor(mx, off, 32));
  float e1 = __expf(a1 - mx), e2 = __expf(a2 - mx);
  float ssum = e1 + e2;
  for (int off=16; off>=1; off>>=1) ssum += __shfl_xor(ssum, off, 32);
  float inv = 1.f / ssum;
  attn[(size_t)bc*64 + lane]      = e1*inv;
  attn[(size_t)bc*64 + lane + 32] = e2*inv;
}

__global__ void k_cam_out(const float* __restrict__ f2, const float* __restrict__ mean,
    const float* __restrict__ attn, const float* __restrict__ gc, float* __restrict__ SC){
  int tid = blockIdx.x*blockDim.x + threadIdx.x;  // 16384
  int b = tid >> 12, n = tid & 4095;
  const float* base = f2 + ((size_t)b << 18);
  const float* at = attn + (size_t)b*64*64;
  const float* mb = mean + (b<<6);
  float accv[64];
#pragma unroll
  for (int c=0;c<64;++c) accv[c]=0.f;
#pragma unroll 1
  for (int d=0; d<64; ++d){
    float fd = base[((size_t)d << 12) + n] - mb[d];
#pragma unroll
    for (int c=0;c<64;++c) accv[c] += at[c*64+d]*fd;
  }
  float g = gc[0];
#pragma unroll
  for (int c=0;c<64;++c)
    SC[(((size_t)b*64 + c) << 12) + n] = g*accv[c];
}

// ---------------- final 1x1 conv (64 -> 2) ------------------------------------
__global__ void k_conv8(const float* __restrict__ fs, const float* __restrict__ w8,
    const float* __restrict__ b8, float* __restrict__ out){
  int tid = blockIdx.x*blockDim.x + threadIdx.x;  // 16384
  int b = tid >> 12, n = tid & 4095;
  const float* fb = fs + ((size_t)b << 18);
  float a0=0.f, a1=0.f;
#pragma unroll 1
  for (int c=0;c<64;++c){
    float f = fb[((size_t)c << 12) + n];
    a0 += w8[c]*f;
    a1 += w8[64+c]*f;
  }
  out[(((size_t)(b*2))   << 12) + n] = a0 + b8[0];
  out[(((size_t)(b*2+1)) << 12) + n] = a1 + b8[1];
}

// ---------------- host orchestration ------------------------------------------
extern "C" void kernel_launch(void* const* d_in, const int* in_sizes, int n_in,
                              void* d_out, int out_size, void* d_ws, size_t ws_size,
                              hipStream_t stream)
{
  const float* x    = (const float*)d_in[0];
  const float* w5a  = (const float*)d_in[1];
  const float* bn5a = (const float*)d_in[2];
  const float* w5c  = (const float*)d_in[3];
  const float* bn5c = (const float*)d_in[4];
  const float* qw   = (const float*)d_in[5];
  const float* qb   = (const float*)d_in[6];
  const float* kw   = (const float*)d_in[7];
  const float* kb_  = (const float*)d_in[8];
  const float* vw   = (const float*)d_in[9];
  const float* vb   = (const float*)d_in[10];
  const float* g_pam= (const float*)d_in[11];
  const float* g_cam= (const float*)d_in[12];
  const float* w51  = (const float*)d_in[13];
  const float* bn51 = (const float*)d_in[14];
  const float* w52  = (const float*)d_in[15];
  const float* bn52 = (const float*)d_in[16];
  const float* w8   = (const float*)d_in[17];
  const float* b8   = (const float*)d_in[18];
  float* out = (float*)d_out;

  char* ws = (char*)d_ws;                       // ~43.8 MB used, all 256B aligned
  _Float16* P   = (_Float16*)(ws + 0);          // patch buffer (reused), 18,874,368 B
  _Float16* A5  = (_Float16*)(ws + 18874368);   // 128x2304 f16
  _Float16* A51 = (_Float16*)(ws + 19464192);   // 64x576 f16
  _Float16* A52 = (_Float16*)(ws + 19537920);
  float* F1  = (float*)(ws + 19611648);         // 4x64x4096 f32 each
  float* F2  = (float*)(ws + 23805952);
  float* SAb = (float*)(ws + 28000256);
  float* SCb = (float*)(ws + 32194560);
  float* FS  = (float*)(ws + 36388864);
  _Float16* QT = (_Float16*)(ws + 40583168);    // [B*N][16] f16 (zero-padded)
  _Float16* KT = (_Float16*)(ws + 41107456);    // [B*N][16] f16 (zero-padded)
  _Float16* Vv = (_Float16*)(ws + 41631744);    // [B][64][4096] f16
  float* MEAN  = (float*)(ws + 43728896);
  float* ATT   = (float*)(ws + 43729920);

  k_pack5<<<1152, 256, 0, stream>>>(w5a, w5c, A5);
  k_packf<<<144, 256, 0, stream>>>(w51, A51, 36864);
  k_packf<<<144, 256, 0, stream>>>(w52, A52, 36864);

  for (int b=0; b<4; ++b){
    k_im2col<<<4096, 256, 0, stream>>>(x, P, b, 256, 8);
    k_gemm<4,0><<<dim3(64,2), 128, 0, stream>>>(A5, P, 2304, b, F1, F2, bn5a, bn5c);
  }
  k_qkv<<<128, 128, 0, stream>>>(F1, qw, qb, kw, kb_, vw, vb, QT, KT, Vv);
  k_flash<<<dim3(256,4), 32, 0, stream>>>(QT, KT, Vv, g_pam, SAb);
  k_cam_mean<<<256, 256, 0, stream>>>(F2, MEAN);
  k_cam_energy<<<256, 32, 0, stream>>>(F2, MEAN, ATT);
  k_cam_out<<<256, 64, 0, stream>>>(F2, MEAN, ATT, g_cam, SCb);
  for (int b=0; b<4; ++b){
    k_im2col<<<1024, 256, 0, stream>>>(SAb, P, b, 64, 6);
    k_gemm<4,1><<<dim3(64,1), 128, 0, stream>>>(A51, P, 576, b, FS, nullptr, bn51, nullptr);
  }
  for (int b=0; b<4; ++b){
    k_im2col<<<1024, 256, 0, stream>>>(SCb, P, b, 64, 6);
    k_gemm<4,2><<<dim3(64,1), 128, 0, stream>>>(A52, P, 576, b, FS, nullptr, bn52, nullptr);
  }
  k_conv8<<<64, 256, 0, stream>>>(FS, w8, b8, out);
}
